// VarianceAdaptor_73272142070081
// MI455X (gfx1250) — compile-verified, matrix operand order checked by reading, not run on hardware
//
#include <hip/hip_runtime.h>
#include <hip/hip_bf16.h>
#include <stdint.h>

#define D     256
#define KW    3
#define NBINS 256

typedef __bf16 bf16;
typedef __attribute__((ext_vector_type(16))) __bf16 v16bf;
typedef __attribute__((ext_vector_type(8)))  float  v8f;

// ---------------------------------------------------------------------------
// Weight repack: w[kw][cin][n] (f32) -> bf16 B-fragments ready for
// v_wmma_f32_16x16x32_bf16.  Fragment f = ((kw*8 + cb)*16 + nb) covers
// K-block cb (32 cin) x N-block nb (16 n).  Lane l holds n = nb*16 + (l&15);
// element i holds k = base + i (i<8) or base + 16 + (i-8), base = (l<16?0:8).
// Stored contiguously per lane -> wave loads 2x b128, fully coalesced.
// ---------------------------------------------------------------------------
__global__ void pack_w_kernel(const float* __restrict__ w, bf16* __restrict__ wp) {
  int e = blockIdx.x * blockDim.x + threadIdx.x;
  if (e >= KW * 8 * 16 * 512) return;
  int f  = e >> 9;
  int r  = e & 511;
  int l  = r >> 4;
  int i  = r & 15;
  int kw = f >> 7;
  int cb = (f >> 4) & 7;
  int nb = f & 15;
  int n    = nb * 16 + (l & 15);
  int base = (l < 16) ? 0 : 8;
  int kk   = (i < 8) ? (base + i) : (base + 16 + (i - 8));
  int cin  = cb * 32 + kk;
  wp[e] = (bf16)w[(kw * D + cin) * D + n];
}

// ---------------------------------------------------------------------------
// Fused conv1d(K=3, SAME) + bias + ReLU + LayerNorm.
// Block: 256 threads = 8 waves (2 M x 4 N), computes 32 rows x 256 cols.
// A window (34 rows incl. halo) staged ONCE via global_load_async_to_lds_b128
// (ASYNCcnt), f32 in LDS, converted to bf16 at fragment read.  kw loop only
// shifts the LDS row.  LDS A window is reused as the LayerNorm buffer.
// ---------------------------------------------------------------------------
__global__ __launch_bounds__(256) void conv_ln_wmma_kernel(
    const float* __restrict__ in, const bf16* __restrict__ wp,
    const float* __restrict__ bias, const float* __restrict__ g,
    const float* __restrict__ be, float* __restrict__ out, int Tlen) {
  __shared__ __align__(16) float S[34][D + 4];   // A window, then LN buffer
  __shared__ float psum[32][8], psq[32][8];
  __shared__ float mean_s[32], rstd_s[32];

  const int tid  = threadIdx.x;
  const int lane = tid & 31;
  const int wave = tid >> 5;
  const int wm   = wave >> 2;   // 0..1 (M sub-tile)
  const int wn   = wave & 3;    // 0..3 (N sub-tile, 64 cols)
  const int m0   = blockIdx.x * 32;

  // Tlen % 32 == 0  ->  whole block lies in one sequence.
  const int b  = m0 / Tlen;
  const int t0 = m0 - b * Tlen;

  // ---- stage rows ts = t0-1 .. t0+32 (f32) via async copy to LDS ----------
  for (int e4 = tid; e4 < 34 * (D / 4); e4 += 256) {
    int lr = e4 >> 6;
    int c4 = (e4 & 63) << 2;
    int ts = t0 - 1 + lr;
    if (ts >= 0 && ts < Tlen) {
      unsigned lds = (unsigned)(uintptr_t)(&S[lr][c4]);
      const float* gp = &in[((long)(b * Tlen + ts)) * D + c4];
      asm volatile("global_load_async_to_lds_b128 %0, %1, off"
                   :: "v"(lds), "v"(gp) : "memory");
    } else {
      S[lr][c4 + 0] = 0.f; S[lr][c4 + 1] = 0.f;
      S[lr][c4 + 2] = 0.f; S[lr][c4 + 3] = 0.f;
    }
  }
  asm volatile("s_wait_asynccnt 0" ::: "memory");
  __syncthreads();

  v8f acc[4] = {};
  const int arow  = wm * 16 + (lane & 15);     // tile-local row
  const int abase = (lane < 16) ? 0 : 8;

  for (int kw = 0; kw < KW; ++kw) {
    for (int cb = 0; cb < 8; ++cb) {
      // A fragment (16x32) from LDS f32 window, convert to bf16.
      v16bf a;
      const float* ap = &S[arow + kw][cb * 32];
#pragma unroll
      for (int i = 0; i < 8; ++i) {
        a[i]     = (bf16)ap[abase + i];
        a[8 + i] = (bf16)ap[abase + 16 + i];
      }
#pragma unroll
      for (int nb = 0; nb < 4; ++nb) {
        int f = (kw * 8 + cb) * 16 + (wn * 4 + nb);
        const bf16* bp = wp + (long)f * 512 + lane * 16;
        v16bf bfr;
#pragma unroll
        for (int i = 0; i < 16; ++i) bfr[i] = bp[i];
        acc[nb] = __builtin_amdgcn_wmma_f32_16x16x32_bf16(
            false, a, false, bfr, (short)0, acc[nb], false, false);
      }
    }
  }

  // ---- bias + ReLU -> LDS (reuse S rows 0..31 as output buffer) -----------
  __syncthreads();   // all A reads done before overwrite
#pragma unroll
  for (int nb = 0; nb < 4; ++nb) {
    int col    = wn * 64 + nb * 16 + (lane & 15);
    float bcol = bias[col];
#pragma unroll
    for (int r = 0; r < 8; ++r) {
      int rl  = wm * 16 + ((lane < 16) ? r : (8 + r));
      float v = acc[nb][r] + bcol;
      S[rl][col] = v > 0.f ? v : 0.f;
    }
  }
  __syncthreads();

  // ---- LayerNorm: 8 threads per row partial-reduce, then finalize ---------
  {
    int row = tid >> 3, sub = tid & 7;
    float s = 0.f, sq = 0.f;
    const float* rp = &S[row][sub * 32];
#pragma unroll
    for (int i = 0; i < 32; ++i) { float v = rp[i]; s += v; sq += v * v; }
    psum[row][sub] = s;
    psq[row][sub]  = sq;
  }
  __syncthreads();
  if ((tid & 7) == 0) {
    int row = tid >> 3;
    float s = 0.f, sq = 0.f;
#pragma unroll
    for (int i = 0; i < 8; ++i) { s += psum[row][i]; sq += psq[row][i]; }
    float mean = s * (1.f / 256.f);
    float var  = sq * (1.f / 256.f) - mean * mean;
    mean_s[row] = mean;
    rstd_s[row] = rsqrtf(var + 1e-5f);
  }
  __syncthreads();

  for (int e = tid; e < 32 * D; e += 256) {
    int row = e >> 8, col = e & 255;
    int gr  = m0 + row;
    float v = (S[row][col] - mean_s[row]) * rstd_s[row] * g[col] + be[col];
    out[(long)gr * D + col] = v;
  }
}

// ---------------------------------------------------------------------------
// D->1 linear: one wave per row, shuffle reduction.
// ---------------------------------------------------------------------------
__global__ __launch_bounds__(256) void rowdot_kernel(
    const float* __restrict__ h, const float* __restrict__ wl,
    const float* __restrict__ bl, float* __restrict__ out, int rows) {
  int lane = threadIdx.x & 31;
  int wave = threadIdx.x >> 5;
  int row  = blockIdx.x * 8 + wave;
  if (row >= rows) return;
  const float* hp = &h[(long)row * D + lane * 8];
  float s = 0.f;
#pragma unroll
  for (int i = 0; i < 8; ++i) s += hp[i] * wl[lane * 8 + i];
#pragma unroll
  for (int off = 16; off > 0; off >>= 1) s += __shfl_down(s, off, 32);
  if (lane == 0) out[row] = s + bl[0];
}

// ---------------------------------------------------------------------------
// Length regulator pieces.
// ---------------------------------------------------------------------------
__global__ void cumsum_kernel(const int* __restrict__ dur, int* __restrict__ cum,
                              int B_, int L_) {
  int b = blockIdx.x * blockDim.x + threadIdx.x;
  if (b >= B_) return;
  int s = 0;
  for (int i = 0; i < L_; ++i) { s += dur[b * L_ + i]; cum[b * L_ + i] = s; }
}

__global__ void token_kernel(const int* __restrict__ cum, int* __restrict__ tok,
                             int B_, int L_, int T_) {
  int idx = blockIdx.x * blockDim.x + threadIdx.x;
  if (idx >= B_ * T_) return;
  int b = idx / T_;
  int t = idx - b * T_;
  const int* c = &cum[b * L_];
  if (t >= c[L_ - 1]) { tok[idx] = -1; return; }
  int lo = 0, hi = L_;            // upper_bound (searchsorted side='right')
  while (lo < hi) { int mid = (lo + hi) >> 1; if (c[mid] > t) hi = mid; else lo = mid + 1; }
  tok[idx] = lo < L_ - 1 ? lo : (L_ - 1);
}

__global__ __launch_bounds__(256) void gather_kernel(
    const float* __restrict__ x, const int* __restrict__ tok,
    float* __restrict__ xe, int L_, int T_) {
  int row = blockIdx.x * 4 + (threadIdx.x >> 6);
  int c4  = (threadIdx.x & 63) << 2;
  int b   = row / T_;
  int k   = tok[row];
  float4 v = make_float4(0.f, 0.f, 0.f, 0.f);
  if (k >= 0) v = *(const float4*)&x[((long)(b * L_ + k)) * D + c4];
  *(float4*)&xe[(long)row * D + c4] = v;
}

// ---------------------------------------------------------------------------
// bucketize(target) (searchsorted side='left') + embedding add.
// ---------------------------------------------------------------------------
__global__ __launch_bounds__(256) void bucket_add_kernel(
    const float* __restrict__ src, const float* __restrict__ target,
    const float* __restrict__ quant, const float* __restrict__ table,
    float* __restrict__ dst, int nq) {
  int row = blockIdx.x * 4 + (threadIdx.x >> 6);
  int c4  = (threadIdx.x & 63) << 2;
  float v = target[row];
  int lo = 0, hi = nq;            // first i with quant[i] >= v
  while (lo < hi) { int mid = (lo + hi) >> 1; if (quant[mid] >= v) hi = mid; else lo = mid + 1; }
  const float4 e = *(const float4*)&table[(long)lo * D + c4];
  float4 s = *(const float4*)&src[(long)row * D + c4];
  s.x += e.x; s.y += e.y; s.z += e.z; s.w += e.w;
  *(float4*)&dst[(long)row * D + c4] = s;
}

__global__ void copydur_kernel(const int* __restrict__ dur, float* __restrict__ o1,
                               float* __restrict__ o2, int n) {
  int i = blockIdx.x * blockDim.x + threadIdx.x;
  if (i < n) { float v = (float)dur[i]; o1[i] = v; o2[i] = v; }
}

// ---------------------------------------------------------------------------
extern "C" void kernel_launch(void* const* d_in, const int* in_sizes, int n_in,
                              void* d_out, int out_size, void* d_ws, size_t ws_size,
                              hipStream_t stream) {
  (void)in_sizes; (void)n_in; (void)out_size; (void)ws_size;
  const int B = 64, L = 512, T = 2048;

  const float* x               = (const float*)d_in[0];
  const float* pitch_target    = (const float*)d_in[1];
  const float* energy_target   = (const float*)d_in[2];
  const int*   duration_target = (const int*)d_in[3];
  // d_in[4] = max_len scalar (== T)
  const float* dp[10]; const float* pp[10]; const float* ep[10];
  for (int i = 0; i < 10; ++i) {
    dp[i] = (const float*)d_in[5 + i];
    pp[i] = (const float*)d_in[15 + i];
    ep[i] = (const float*)d_in[25 + i];
  }
  const float* pitch_table  = (const float*)d_in[35];
  const float* energy_table = (const float*)d_in[36];
  const float* pitch_quant  = (const float*)d_in[37];
  const float* energy_quant = (const float*)d_in[38];

  // d_out: [xe BxTxD][pitch BxT][energy BxT][logdur BxL][dur BxL][mel_len BxL]
  float* out_xe     = (float*)d_out;
  float* out_pitch  = out_xe + (size_t)B * T * D;
  float* out_energy = out_pitch + (size_t)B * T;
  float* out_logdur = out_energy + (size_t)B * T;
  float* out_dur    = out_logdur + (size_t)B * L;
  float* out_mellen = out_dur + (size_t)B * L;

  char* ws   = (char*)d_ws;
  float* XE  = (float*)ws; ws += (size_t)B * T * D * 4;
  float* T1  = (float*)ws; ws += (size_t)B * T * D * 4;
  float* T2  = (float*)ws; ws += (size_t)B * T * D * 4;
  bf16*  WP  = (bf16*)ws;  ws += (size_t)6 * KW * D * D * 2;
  int*   CUM = (int*)ws;   ws += (size_t)B * L * 4;
  int*   TOK = (int*)ws;

  const int wElems = KW * D * D;  // 196608 bf16 per conv
  const float* wsrc[6] = { dp[0], dp[4], pp[0], pp[4], ep[0], ep[4] };
  for (int i = 0; i < 6; ++i)
    pack_w_kernel<<<(wElems + 255) / 256, 256, 0, stream>>>(wsrc[i], WP + (size_t)i * wElems);

  // duration predictor on x [B,L,D]
  conv_ln_wmma_kernel<<<B * L / 32, 256, 0, stream>>>(x,  WP + 0 * (size_t)wElems, dp[1], dp[2], dp[3], T1, L);
  conv_ln_wmma_kernel<<<B * L / 32, 256, 0, stream>>>(T1, WP + 1 * (size_t)wElems, dp[5], dp[6], dp[7], T2, L);
  rowdot_kernel<<<B * L / 8, 256, 0, stream>>>(T2, dp[8], dp[9], out_logdur, B * L);
  copydur_kernel<<<(B * L + 255) / 256, 256, 0, stream>>>(duration_target, out_dur, out_mellen, B * L);

  // length regulate -> XE [B,T,D]
  cumsum_kernel<<<1, 64, 0, stream>>>(duration_target, CUM, B, L);
  token_kernel<<<(B * T + 255) / 256, 256, 0, stream>>>(CUM, TOK, B, L, T);
  gather_kernel<<<B * T / 4, 256, 0, stream>>>(x, TOK, XE, L, T);

  // pitch predictor on XE, then XE += pitch_emb (in place)
  conv_ln_wmma_kernel<<<B * T / 32, 256, 0, stream>>>(XE, WP + 2 * (size_t)wElems, pp[1], pp[2], pp[3], T1, T);
  conv_ln_wmma_kernel<<<B * T / 32, 256, 0, stream>>>(T1, WP + 3 * (size_t)wElems, pp[5], pp[6], pp[7], T2, T);
  rowdot_kernel<<<B * T / 8, 256, 0, stream>>>(T2, pp[8], pp[9], out_pitch, B * T);
  bucket_add_kernel<<<B * T / 4, 256, 0, stream>>>(XE, pitch_target, pitch_quant, pitch_table, XE, NBINS - 1);

  // energy predictor on XE, then out_xe = XE + energy_emb
  conv_ln_wmma_kernel<<<B * T / 32, 256, 0, stream>>>(XE, WP + 4 * (size_t)wElems, ep[1], ep[2], ep[3], T1, T);
  conv_ln_wmma_kernel<<<B * T / 32, 256, 0, stream>>>(T1, WP + 5 * (size_t)wElems, ep[5], ep[6], ep[7], T2, T);
  rowdot_kernel<<<B * T / 8, 256, 0, stream>>>(T2, ep[8], ep[9], out_energy, B * T);
  bucket_add_kernel<<<B * T / 4, 256, 0, stream>>>(XE, energy_target, energy_quant, energy_table, out_xe, NBINS - 1);
}
